// SEALModel_70806830841999
// MI455X (gfx1250) — matrix operand order repaired and checked by reading, not compiled, not run on hardware
//
#include <hip/hip_runtime.h>

// ---------------------------------------------------------------------------
// SEAL GNN forward for MI455X (gfx1250, wave32, WMMA).
//   N=50000 nodes, E=1.25M edges, G=1024 graphs, LABEL_DIM=32, HIDDEN=64.
// Memory-bound on edge scatter (≈1 GB traffic -> ~45us @ 23.3 TB/s); GEMMs
// (~1.4 GFLOP) run on v_wmma_f32_16x16x32_f16.
// ---------------------------------------------------------------------------

typedef _Float16 v16h __attribute__((ext_vector_type(16)));
typedef float    v8f  __attribute__((ext_vector_type(8)));

#define MAX_LABEL 50

// ---------------- embedding gather: h[i,0:32] = emb[clip(label[i])] --------
__global__ void embed_kernel(const int* __restrict__ labels,
                             const float* __restrict__ emb,
                             float* __restrict__ h, int n) {
  int t = blockIdx.x * blockDim.x + threadIdx.x;
  if (t >= n * 32) return;
  int i = t >> 5, c = t & 31;
  int lab = labels[i];
  lab = lab < 0 ? 0 : (lab > MAX_LABEL ? MAX_LABEL : lab);
  h[(size_t)i * 32 + c] = emb[(size_t)lab * 32 + c];
}

// ---------------- zero fill ------------------------------------------------
__global__ void zero_kernel(float* __restrict__ p, int n) {
  int t = blockIdx.x * blockDim.x + threadIdx.x;
  if (t < n) p[t] = 0.0f;
}

// ---------------- edge scatter: agg[dst] += h[src] (float4 + HW f32 atomics)
template <int DIN>
__global__ void scatter_kernel(const float* __restrict__ h,
                               float* __restrict__ agg,
                               const int* __restrict__ src,
                               const int* __restrict__ dst, int e_count) {
  constexpr int QP = DIN / 4;                 // float4 chunks per edge
  int t = blockIdx.x * blockDim.x + threadIdx.x;
  if (t >= e_count * QP) return;
  int e = t / QP;
  int q = (t % QP) * 4;
  int s = src[e], d = dst[e];
  const float4 v = *reinterpret_cast<const float4*>(h + (size_t)s * DIN + q);
  float* ap = agg + (size_t)d * DIN + q;
  unsafeAtomicAdd(ap + 0, v.x);               // global_atomic_add_f32
  unsafeAtomicAdd(ap + 1, v.y);
  unsafeAtomicAdd(ap + 2, v.z);
  unsafeAtomicAdd(ap + 3, v.w);
}

// ---------------- GIN MLP:  hout = relu(relu((h+agg)@W1+b1)@W2+b2) ---------
// One wave handles one 16-row tile, all 64 output cols (4 col-tiles of 16).
// WMMA v_wmma_f32_16x16x32_f16; inter-GEMM relayout via per-wave LDS tile.
template <int DIN>
__global__ __launch_bounds__(128) void gin_mlp_kernel(
    const float* __restrict__ hin, const float* __restrict__ agg,
    const float* __restrict__ w1, const float* __restrict__ b1,
    const float* __restrict__ w2, const float* __restrict__ b2,
    float* __restrict__ hout, int n) {
  __shared__ float lds_t[4][16 * 64];          // 4 waves/block * 4KB
  const int lane = threadIdx.x & 31;
  const int wv   = threadIdx.x >> 5;
  const int tile = blockIdx.x * 4 + wv;
  const int row0 = tile * 16;
  const int m    = lane & 15;                  // A: row M ; B/C: col N
  const int hi   = lane >> 4;                  // lane half select
  const int row  = row0 + m;
  const bool rv  = row < n;
  constexpr int KC1 = DIN / 32;                // K chunks for GEMM1

  // ---- A fragments for GEMM1 (z = h + agg), ISA 16-bit 16x32 A layout:
  // lanes 0-15: g<4 -> K=2g..2g+1, g>=4 -> K=2g+8..; lanes 16-31: +8.
  v16h aF[KC1];
#pragma unroll
  for (int kc = 0; kc < KC1; ++kc) {
#pragma unroll
    for (int e2 = 0; e2 < 16; ++e2) {
      int g = e2 >> 1, p = e2 & 1;
      int kl = 2 * g + p + 8 * hi + ((g >= 4) ? 8 : 0);
      float v = 0.0f;
      if (rv) {
        size_t idx = (size_t)row * DIN + kc * 32 + kl;
        v = hin[idx] + agg[idx];
      }
      aF[kc][e2] = (_Float16)v;
    }
  }

  // ---- GEMM1: t = relu(z @ W1 + b1) -> LDS tile [16][64]
#pragma unroll
  for (int nt = 0; nt < 4; ++nt) {
    const int n0 = nt * 16;
    v8f c;
#pragma unroll
    for (int r = 0; r < 8; ++r) c[r] = b1[n0 + m];
#pragma unroll
    for (int kc = 0; kc < KC1; ++kc) {
      // B 32x16 layout: same K order as A (lanes 0-15 K=0..7,16..23; lanes 16-31 +8)
      v16h bF;
#pragma unroll
      for (int e2 = 0; e2 < 16; ++e2) {
        int g = e2 >> 1, p = e2 & 1;
        int kl = 2 * g + p + 8 * hi + ((g >= 4) ? 8 : 0);
        bF[e2] = (_Float16)w1[(size_t)(kc * 32 + kl) * 64 + n0 + m];
      }
      c = __builtin_amdgcn_wmma_f32_16x16x32_f16(false, aF[kc], false, bF,
                                                 (short)0, c, false, false);
    }
    // C layout: element (M = r + 8*hi, N = n0 + m)
#pragma unroll
    for (int r = 0; r < 8; ++r)
      lds_t[wv][(r + 8 * hi) * 64 + n0 + m] = fmaxf(c[r], 0.0f);
  }

  // per-wave private LDS region: just drain DS ops (CDNA5 split counter)
  asm volatile("s_wait_dscnt 0" ::: "memory");

  // ---- A fragments for GEMM2 from LDS tile (K = 64 -> 2 chunks)
  v16h a2[2];
#pragma unroll
  for (int kc = 0; kc < 2; ++kc) {
#pragma unroll
    for (int e2 = 0; e2 < 16; ++e2) {
      int g = e2 >> 1, p = e2 & 1;
      int kl = 2 * g + p + 8 * hi + ((g >= 4) ? 8 : 0);
      a2[kc][e2] = (_Float16)lds_t[wv][m * 64 + kc * 32 + kl];
    }
  }

  // ---- GEMM2: hout = relu(t @ W2 + b2)
#pragma unroll
  for (int nt = 0; nt < 4; ++nt) {
    const int n0 = nt * 16;
    v8f c;
#pragma unroll
    for (int r = 0; r < 8; ++r) c[r] = b2[n0 + m];
#pragma unroll
    for (int kc = 0; kc < 2; ++kc) {
      v16h bF;
#pragma unroll
      for (int e2 = 0; e2 < 16; ++e2) {
        int g = e2 >> 1, p = e2 & 1;
        int kl = 2 * g + p + 8 * hi + ((g >= 4) ? 8 : 0);
        bF[e2] = (_Float16)w2[(size_t)(kc * 32 + kl) * 64 + n0 + m];
      }
      c = __builtin_amdgcn_wmma_f32_16x16x32_f16(false, a2[kc], false, bF,
                                                 (short)0, c, false, false);
    }
#pragma unroll
    for (int r = 0; r < 8; ++r) {
      int orow = row0 + r + 8 * hi;
      if (orow < n) hout[(size_t)orow * 64 + n0 + m] = fmaxf(c[r], 0.0f);
    }
  }
}

// ---------------- per-graph pooling ---------------------------------------
__global__ void pool_cnt_kernel(const int* __restrict__ gid,
                                float* __restrict__ cnts, int n) {
  int i = blockIdx.x * blockDim.x + threadIdx.x;
  if (i >= n) return;
  unsafeAtomicAdd(&cnts[gid[i]], 1.0f);
}

__global__ void pool_sum_kernel(const float* __restrict__ h,
                                const int* __restrict__ gid,
                                float* __restrict__ sums, int n) {
  int t = blockIdx.x * blockDim.x + threadIdx.x;
  if (t >= n * 16) return;
  int i = t >> 4;
  int q = (t & 15) * 4;
  const float4 v = *reinterpret_cast<const float4*>(h + (size_t)i * 64 + q);
  float* sp = sums + (size_t)gid[i] * 64 + q;
  unsafeAtomicAdd(sp + 0, v.x);
  unsafeAtomicAdd(sp + 1, v.y);
  unsafeAtomicAdd(sp + 2, v.z);
  unsafeAtomicAdd(sp + 3, v.w);
}

// ---------------- scorer MLP: [G,64] -> relu(@sw1+sb1) @ sw2 + sb2 --------
__global__ void scorer_kernel(const float* __restrict__ sums,
                              const float* __restrict__ cnts,
                              const float* __restrict__ sw1,
                              const float* __restrict__ sb1,
                              const float* __restrict__ sw2,
                              const float* __restrict__ sb2,
                              float* __restrict__ out, int g_count) {
  int g = blockIdx.x * blockDim.x + threadIdx.x;
  if (g >= g_count) return;
  float inv = 1.0f / fmaxf(cnts[g], 1.0f);
  float hg[64];
#pragma unroll 8
  for (int c = 0; c < 64; ++c) hg[c] = sums[(size_t)g * 64 + c] * inv;
  float acc = sb2[0];
  for (int j = 0; j < 64; ++j) {
    float t = sb1[j];
#pragma unroll 8
    for (int c = 0; c < 64; ++c) t += hg[c] * sw1[(size_t)c * 64 + j];
    acc += fmaxf(t, 0.0f) * sw2[j];
  }
  out[g] = acc;
}

// ---------------------------------------------------------------------------
extern "C" void kernel_launch(void* const* d_in, const int* in_sizes, int n_in,
                              void* d_out, int out_size, void* d_ws,
                              size_t ws_size, hipStream_t stream) {
  const int*   labels = (const int*)d_in[0];
  const int*   src    = (const int*)d_in[1];
  const int*   dst    = (const int*)d_in[2];
  const int*   gid    = (const int*)d_in[3];
  const float* emb    = (const float*)d_in[4];
  const float* w1_0   = (const float*)d_in[5];
  const float* b1_0   = (const float*)d_in[6];
  const float* w2_0   = (const float*)d_in[7];
  const float* b2_0   = (const float*)d_in[8];
  const float* w1_1   = (const float*)d_in[9];
  const float* b1_1   = (const float*)d_in[10];
  const float* w2_1   = (const float*)d_in[11];
  const float* b2_1   = (const float*)d_in[12];
  const float* sw1    = (const float*)d_in[13];
  const float* sb1    = (const float*)d_in[14];
  const float* sw2    = (const float*)d_in[15];
  const float* sb2    = (const float*)d_in[16];

  const int n = in_sizes[0];      // N nodes
  const int e = in_sizes[1];      // E edges
  const int G = out_size;         // graphs (output is [G,1])

  // workspace layout (floats)
  float* ws   = (float*)d_ws;
  float* hA   = ws;                             // [N,64] (embed uses [N,32])
  float* hB   = ws + (size_t)n * 64;            // [N,64]
  float* agg  = ws + 2 * (size_t)n * 64;        // [N,64]
  float* sums = ws + 3 * (size_t)n * 64;        // [G,64]
  float* cnts = sums + (size_t)G * 64;          // [G]

  const int T = 256;
  auto cdiv = [](int a, int b) { return (a + b - 1) / b; };
  const int tiles  = cdiv(n, 16);
  const int mblks  = cdiv(tiles, 4);            // 4 waves (tiles) per block

  // 1) embedding
  embed_kernel<<<cdiv(n * 32, T), T, 0, stream>>>(labels, emb, hA, n);

  // 2) GIN layer 0 (din=32)
  zero_kernel<<<cdiv(n * 32, T), T, 0, stream>>>(agg, n * 32);
  scatter_kernel<32><<<cdiv(e * 8, T), T, 0, stream>>>(hA, agg, src, dst, e);
  gin_mlp_kernel<32><<<mblks, 128, 0, stream>>>(hA, agg, w1_0, b1_0, w2_0,
                                                b2_0, hB, n);

  // 3) GIN layer 1 (din=64)
  zero_kernel<<<cdiv(n * 64, T), T, 0, stream>>>(agg, n * 64);
  scatter_kernel<64><<<cdiv(e * 16, T), T, 0, stream>>>(hB, agg, src, dst, e);
  gin_mlp_kernel<64><<<mblks, 128, 0, stream>>>(hB, agg, w1_1, b1_1, w2_1,
                                                b2_1, hA, n);

  // 4) mean pool
  zero_kernel<<<cdiv(G * 65, T), T, 0, stream>>>(sums, G * 65);
  pool_cnt_kernel<<<cdiv(n, T), T, 0, stream>>>(gid, cnts, n);
  pool_sum_kernel<<<cdiv(n * 16, T), T, 0, stream>>>(hA, gid, sums, n);

  // 5) scorer
  scorer_kernel<<<cdiv(G, 64), 64, 0, stream>>>(sums, cnts, sw1, sb1, sw2,
                                                sb2, (float*)d_out, G);
}
